// GraphRecModel_89249420411232
// MI455X (gfx1250) — matrix operand order repaired and checked by reading, not compiled, not run on hardware
//
#include <hip/hip_runtime.h>
#include <hip/hip_bf16.h>

// ---------------------------------------------------------------------------
// GraphRec fused pipeline for MI455X (gfx1250, wave32, WMMA + TDM).
//
// Stages:
//   1) proj_kernel      : fs = feat_src@W_src+b ; fd = feat_dst@W_dst+b   (WMMA)
//   2) edge_kernel      : fused 5-layer per-edge MLP chain with gathers,
//                         ending in logits[E] = (lrelu(...)@W_el+b_el)·attn (WMMA)
//   3) max/exp kernels  : segment softmax over dst via L2 atomics
//   4) agg_kernel       : out[dst] += fs[src] * a   (float atomics, L2)
//
// Tile staging (global 16x64 f32 -> padded LDS) is done by the Tensor Data
// Mover (tensor_load_to_lds, TENSORcnt), with D#.pad_* reproducing the
// A_STRIDE=68 bank padding, so the DMA overlaps weight staging and no VGPRs
// are spent on the copy.
// ---------------------------------------------------------------------------

#define DD 64
#define NEGS 0.2f
#define W_STRIDE 72      // f16 elements per N-major weight row in LDS (pad vs banks)
#define A_STRIDE 68      // f32 elements per activation row in LDS (= 64 + TDM pad 4)
#define EDGE_WAVES 4
#define PROJ_WAVES 8

typedef __attribute__((ext_vector_type(16))) _Float16 v16h;
typedef __attribute__((ext_vector_type(8)))  float    v8f;
typedef __attribute__((ext_vector_type(4)))  unsigned u32x4;
typedef __attribute__((ext_vector_type(8)))  unsigned u32x8;

__device__ __forceinline__ v8f wmma_f16(v16h a, v16h b, v8f c) {
  // D = A(16x32 f16) * B(32x16 f16) + C(16x16 f32)
  return __builtin_amdgcn_wmma_f32_16x16x32_f16(false, a, false, b, (short)0, c,
                                                false, false);
}

// ---------------------------------------------------------------------------
// TDM: DMA one 16-row x 64-col f32 tile (rows contiguous in global) into LDS
// at byte offset lds_off, inserting 4 DWORDs of padding after every 64 DWORDs
// (-> rows land A_STRIDE=68 floats apart).  D# per CDNA5 ISA.
// ---------------------------------------------------------------------------
__device__ __forceinline__ void tdm_load_tile_16x64_f32(unsigned lds_off,
                                                        const float* gsrc) {
  const unsigned long long ga = (unsigned long long)(uintptr_t)gsrc;
  u32x4 g0;
  g0.x = 1u;                                 // count=1 (valid), no gather/restore
  g0.y = lds_off;                            // lds_addr (bytes)
  g0.z = (unsigned)ga;                       // global_addr[31:0]
  g0.w = (unsigned)(ga >> 32) | (2u << 30);  // global_addr[56:32] | type=2 (image)
  u32x8 g1;
  g1[0] = (2u << 16)      // data_size = 4 B
        | (1u << 20)      // pad_enable
        | (5u << 22)      // pad_interval code 5 = every 64 DWORDs
        | (3u << 25);     // pad_amount  code 3 = 4 DWORDs
  g1[1] = (64u & 0xFFFFu) << 16;                    // tensor_dim0 = 64 (lo 16)
  g1[2] = (64u >> 16) | ((0x7FFFFFFFu & 0xFFFFu) << 16);  // dim0 hi | tensor_dim1 lo
  g1[3] = (0x7FFFFFFFu >> 16) | (64u << 16);        // tensor_dim1 hi | tile_dim0=64
  g1[4] = 16u;                                      // tile_dim1=16, tile_dim2=0
  g1[5] = 64u;                                      // tensor_dim0_stride = 64
  g1[6] = 0u;                                       // stride0 hi | stride1 lo
  g1[7] = 0u;                                       // stride1 hi
  // 2-group form: VADDR2/VADDR3 = NULL (2D tensor)
  asm volatile("tensor_load_to_lds %0, %1" :: "s"(g0), "s"(g1) : "memory");
}

// A fragment: lane l holds row m=l&15; half=l>>4 selects K sub-bands.
// v16h element pairs (VGPR j): K = 2j{+1} + 16*half + kbase
__device__ __forceinline__ v16h load_a_frag(const float* act, int lane, int kbase,
                                            bool do_lrelu) {
  const int m = lane & 15, half = lane >> 4;
  const float* row = act + m * A_STRIDE + kbase;
  v16h a;
#pragma unroll
  for (int j = 0; j < 8; ++j) {
    const int k = 2 * j + 16 * half;
    float x0 = row[k], x1 = row[k + 1];
    if (do_lrelu) {
      x0 = x0 >= 0.f ? x0 : NEGS * x0;
      x1 = x1 >= 0.f ? x1 : NEGS * x1;
    }
    a[2 * j]     = (_Float16)x0;
    a[2 * j + 1] = (_Float16)x1;
  }
  return a;
}

// B fragment: lane l holds column n = (l&15)+16*ntile; K = 2j{+1} + 16*half + kbase.
// Weights stored N-major in LDS: w[n*W_STRIDE + k], so K pairs are contiguous.
__device__ __forceinline__ v16h load_b_frag(const _Float16* w, int lane, int ntile,
                                            int kbase) {
  const int n = (lane & 15) + 16 * ntile, half = lane >> 4;
  const _Float16* row = w + n * W_STRIDE + kbase + 16 * half;
  v16h b;
#pragma unroll
  for (int j = 0; j < 8; ++j) {
    b[2 * j]     = row[2 * j];
    b[2 * j + 1] = row[2 * j + 1];
  }
  return b;
}

// One 16x64 = (16x64)@(64x64) layer: 4 N-tiles x 2 K-steps of WMMA.
__device__ __forceinline__ void gemm64(const float* act, const _Float16* w, int lane,
                                       bool do_lrelu, v8f acc[4]) {
  const v16h a0 = load_a_frag(act, lane, 0,  do_lrelu);
  const v16h a1 = load_a_frag(act, lane, 32, do_lrelu);
#pragma unroll
  for (int t = 0; t < 4; ++t) {
    v8f c = {};
    c = wmma_f16(a0, load_b_frag(w, lane, t, 0),  c);
    c = wmma_f16(a1, load_b_frag(w, lane, t, 32), c);
    acc[t] = c;
  }
}

// C/D layout: VGPR r, lane l -> element (m = r + 8*(l>>4), n = (l&15) + 16*t).
// Epilogue: act[m][n] = acc + bias[n] (+ gather[rows[m]*64 + n]).
__device__ __forceinline__ void store_acc(float* act, int lane, const v8f acc[4],
                                          const float* __restrict__ bias,
                                          const float* __restrict__ gbase,
                                          const int* ridx) {
  const int half = lane >> 4, nl = lane & 15;
  int rows[8] = {0, 0, 0, 0, 0, 0, 0, 0};
  if (gbase) {
#pragma unroll
    for (int r = 0; r < 8; ++r) rows[r] = ridx[r + 8 * half];
  }
#pragma unroll
  for (int t = 0; t < 4; ++t) {
    const int n = nl + 16 * t;
    const float b = bias[n];
#pragma unroll
    for (int r = 0; r < 8; ++r) {
      const int m = r + 8 * half;
      float v = acc[t][r] + b;
      if (gbase) v += gbase[(size_t)rows[r] * DD + n];
      act[m * A_STRIDE + n] = v;
    }
  }
}

// ---------------------------------------------------------------------------
// Node projection: out[M x 64] = feat[M x 64] @ W + b  (one 16-row tile / wave)
// ---------------------------------------------------------------------------
__global__ __launch_bounds__(PROJ_WAVES * 32) void proj_kernel(
    const float* __restrict__ feat, const float* __restrict__ W,
    const float* __restrict__ b, float* __restrict__ out, int n_tiles) {
  __shared__ _Float16 s_w[DD * W_STRIDE];
  __shared__ float s_act[PROJ_WAVES][16 * A_STRIDE];

  const int tid = threadIdx.x, lane = tid & 31, wave = tid >> 5;
  const int tile = blockIdx.x * PROJ_WAVES + wave;
  const bool active = tile < n_tiles;
  const long base = (long)tile * 16;
  float* act = s_act[wave];

  if (active)  // kick DMA early so it overlaps the weight staging below
    tdm_load_tile_16x64_f32((unsigned)(uintptr_t)act, feat + base * DD);

  for (int i = tid; i < DD * DD; i += blockDim.x) {
    const int k = i >> 6, n = i & 63;           // W is [k][n] row-major
    s_w[n * W_STRIDE + k] = (_Float16)W[i];     // store N-major
  }
  __syncthreads();
  if (!active) return;

  __builtin_amdgcn_s_wait_tensorcnt(0);

  v8f acc[4];
  gemm64(act, s_w, lane, false, acc);

  const int half = lane >> 4, nl = lane & 15;
#pragma unroll
  for (int t = 0; t < 4; ++t) {
    const int n = nl + 16 * t;
    const float bb = b[n];
#pragma unroll
    for (int r = 0; r < 8; ++r)
      out[(base + r + 8 * half) * DD + n] = acc[t][r] + bb;
  }
}

// ---------------------------------------------------------------------------
// Fused per-edge chain: fe=e_feat@W_e+b_e; ef=fe+fs[src];
// ef=lrelu(ef)@W_m{1,2,3}+b; e=lrelu(ef+fd[dst]); e=e@W_el+b_el; logit=e.attn
// ---------------------------------------------------------------------------
__global__ __launch_bounds__(EDGE_WAVES * 32) void edge_kernel(
    const float* __restrict__ e_feat, const float* __restrict__ fs,
    const float* __restrict__ fd, const int* __restrict__ src_idx,
    const int* __restrict__ dst_idx, const float* __restrict__ W_e,
    const float* __restrict__ b_e, const float* __restrict__ W_m1,
    const float* __restrict__ b_m1, const float* __restrict__ W_m2,
    const float* __restrict__ b_m2, const float* __restrict__ W_m3,
    const float* __restrict__ b_m3, const float* __restrict__ W_el,
    const float* __restrict__ b_el, const float* __restrict__ attn,
    float* __restrict__ logits, int n_tiles) {
  __shared__ _Float16 s_w[5][DD * W_STRIDE];               // 46080 B
  __shared__ float s_act[EDGE_WAVES][16 * A_STRIDE];       // 17408 B
  __shared__ int s_idx[EDGE_WAVES][2][16];                 //   512 B

  const int tid = threadIdx.x, lane = tid & 31, wave = tid >> 5;
  const int tile = blockIdx.x * EDGE_WAVES + wave;
  const bool active = tile < n_tiles;
  const long base_e = (long)tile * 16;
  float* act = s_act[wave];

  if (active)  // DMA the e_feat tile; overlaps the weight staging below
    tdm_load_tile_16x64_f32((unsigned)(uintptr_t)act, e_feat + base_e * DD);

  const float* Wg[5] = {W_e, W_m1, W_m2, W_m3, W_el};
#pragma unroll
  for (int w5 = 0; w5 < 5; ++w5)
    for (int i = tid; i < DD * DD; i += blockDim.x) {
      const int k = i >> 6, n = i & 63;
      s_w[w5][n * W_STRIDE + k] = (_Float16)Wg[w5][i];
    }
  __syncthreads();
  if (!active) return;

  if (lane < 16) {
    s_idx[wave][0][lane] = src_idx[base_e + lane];
    s_idx[wave][1][lane] = dst_idx[base_e + lane];
  }
  __builtin_amdgcn_s_wait_tensorcnt(0);   // e_feat tile resident in LDS
  // All act traffic below is wave-private LDS: DS ops are in-order per wave,
  // so no workgroup barrier is needed between layers.

  v8f acc[4];
  gemm64(act, s_w[0], lane, false, acc);                       // fe = e_feat@W_e
  store_acc(act, lane, acc, b_e, fs, s_idx[wave][0]);          // + b_e + fs[src]
  gemm64(act, s_w[1], lane, true, acc);                        // lrelu -> @W_m1
  store_acc(act, lane, acc, b_m1, nullptr, nullptr);
  gemm64(act, s_w[2], lane, true, acc);                        // lrelu -> @W_m2
  store_acc(act, lane, acc, b_m2, nullptr, nullptr);
  gemm64(act, s_w[3], lane, true, acc);                        // lrelu -> @W_m3
  store_acc(act, lane, acc, b_m3, fd, s_idx[wave][1]);         // + b_m3 + fd[dst]
  gemm64(act, s_w[4], lane, true, acc);                        // lrelu -> @W_el

  // logits: e.attn, reduced over n (lanes within 16-lane half + 4 N-tiles)
  {
    const int half = lane >> 4, nl = lane & 15;
    float q[8] = {0, 0, 0, 0, 0, 0, 0, 0};
#pragma unroll
    for (int t = 0; t < 4; ++t) {
      const int n = nl + 16 * t;
      const float an = attn[n], bb = b_el[n];
#pragma unroll
      for (int r = 0; r < 8; ++r) q[r] += (acc[t][r] + bb) * an;
    }
#pragma unroll
    for (int off = 1; off < 16; off <<= 1) {
#pragma unroll
      for (int r = 0; r < 8; ++r) q[r] += __shfl_xor(q[r], off, 32);
    }
    if (nl == 0) {
#pragma unroll
      for (int r = 0; r < 8; ++r) logits[base_e + r + 8 * half] = q[r];
    }
  }
}

// ---------------------------------------------------------------------------
// Segment softmax + aggregation (L2 atomics; dst tables are L2-resident)
// ---------------------------------------------------------------------------
__device__ __forceinline__ unsigned enc_f(float f) {
  const unsigned u = __float_as_uint(f);
  return (u & 0x80000000u) ? ~u : (u | 0x80000000u);   // order-preserving map
}
__device__ __forceinline__ float dec_f(unsigned u) {
  return (u & 0x80000000u) ? __uint_as_float(u & 0x7FFFFFFFu) : __uint_as_float(~u);
}

__global__ void init_kernel(float* out, unsigned* mx, float* den, long n_out,
                            int n_dst) {
  const long i = (long)blockIdx.x * blockDim.x + threadIdx.x;
  if (i < n_out) out[i] = 0.f;
  if (i < n_dst) {
    mx[i] = 0u;   // encodes below any real value
    den[i] = 0.f;
  }
}

__global__ void max_kernel(const float* __restrict__ logits,
                           const int* __restrict__ dst, unsigned* __restrict__ mx,
                           long E_) {
  const long e = (long)blockIdx.x * blockDim.x + threadIdx.x;
  if (e >= E_) return;
  atomicMax(&mx[dst[e]], enc_f(logits[e]));
}

__global__ void exp_kernel(const float* __restrict__ logits,
                           const int* __restrict__ dst,
                           const unsigned* __restrict__ mx, float* __restrict__ ex,
                           float* __restrict__ den, long E_) {
  const long e = (long)blockIdx.x * blockDim.x + threadIdx.x;
  if (e >= E_) return;
  const int d = dst[e];
  const float v = __expf(logits[e] - dec_f(mx[d]));
  ex[e] = v;
  atomicAdd(&den[d], v);
}

__global__ void agg_kernel(const float* __restrict__ fs, const float* __restrict__ ex,
                           const float* __restrict__ den,
                           const int* __restrict__ src, const int* __restrict__ dst,
                           float* __restrict__ out, long E_) {
  const long t = (long)blockIdx.x * blockDim.x + threadIdx.x;
  const long e = t >> 4;
  if (e >= E_) return;
  const int n4 = (int)(t & 15) * 4;
  const int s = src[e], d = dst[e];
  const float a = ex[e] / den[d];
  const float4 f = *(const float4*)&fs[(long)s * DD + n4];
  float* o = &out[(long)d * DD + n4];
  atomicAdd(&o[0], f.x * a);
  atomicAdd(&o[1], f.y * a);
  atomicAdd(&o[2], f.z * a);
  atomicAdd(&o[3], f.w * a);
}

// ---------------------------------------------------------------------------
extern "C" void kernel_launch(void* const* d_in, const int* in_sizes, int n_in,
                              void* d_out, int out_size, void* d_ws, size_t ws_size,
                              hipStream_t stream) {
  const float* feat_src = (const float*)d_in[0];
  const float* feat_dst = (const float*)d_in[1];
  const float* e_feat   = (const float*)d_in[2];
  const float* W_src = (const float*)d_in[3];
  const float* b_src = (const float*)d_in[4];
  const float* W_dst = (const float*)d_in[5];
  const float* b_dst = (const float*)d_in[6];
  const float* W_e   = (const float*)d_in[7];
  const float* b_e   = (const float*)d_in[8];
  const float* W_m1  = (const float*)d_in[9];
  const float* b_m1  = (const float*)d_in[10];
  const float* W_m2  = (const float*)d_in[11];
  const float* b_m2  = (const float*)d_in[12];
  const float* W_m3  = (const float*)d_in[13];
  const float* b_m3  = (const float*)d_in[14];
  const float* W_el  = (const float*)d_in[15];
  const float* b_el  = (const float*)d_in[16];
  const float* attn  = (const float*)d_in[17];
  const int* src_idx = (const int*)d_in[18];
  const int* dst_idx = (const int*)d_in[19];

  const int n_src = in_sizes[0] / DD;
  const int n_dst = in_sizes[1] / DD;
  const long E_ = in_sizes[18];
  float* out = (float*)d_out;

  // workspace carve-up (f32 units)
  float* ws = (float*)d_ws;
  float* fs = ws;                ws += (long)n_src * DD;
  float* fd = ws;                ws += (long)n_dst * DD;
  float* logits = ws;            ws += E_;
  float* ex = ws;                ws += E_;
  unsigned* mx = (unsigned*)ws;  ws += n_dst;
  float* den = ws;               (void)ws;
  (void)ws_size; (void)n_in; (void)out_size;

  {  // zero output / softmax state every call (harness poisons once)
    const long n_out = (long)n_dst * DD;
    init_kernel<<<(int)((n_out + 255) / 256), 256, 0, stream>>>(out, mx, den,
                                                                n_out, n_dst);
  }
  {  // node projections (WMMA)
    const int ts = n_src / 16;
    proj_kernel<<<(ts + PROJ_WAVES - 1) / PROJ_WAVES, PROJ_WAVES * 32, 0, stream>>>(
        feat_src, W_src, b_src, fs, ts);
    const int td = n_dst / 16;
    proj_kernel<<<(td + PROJ_WAVES - 1) / PROJ_WAVES, PROJ_WAVES * 32, 0, stream>>>(
        feat_dst, W_dst, b_dst, fd, td);
  }
  {  // fused per-edge chain (WMMA)
    const int tiles = (int)(E_ / 16);
    edge_kernel<<<(tiles + EDGE_WAVES - 1) / EDGE_WAVES, EDGE_WAVES * 32, 0,
                  stream>>>(e_feat, fs, fd, src_idx, dst_idx, W_e, b_e, W_m1, b_m1,
                            W_m2, b_m2, W_m3, b_m3, W_el, b_el, attn, logits, tiles);
  }
  const int eb = (int)((E_ + 255) / 256);
  max_kernel<<<eb, 256, 0, stream>>>(logits, dst_idx, mx, E_);
  exp_kernel<<<eb, 256, 0, stream>>>(logits, dst_idx, mx, ex, den, E_);
  agg_kernel<<<(int)((E_ * 16 + 255) / 256), 256, 0, stream>>>(fs, ex, den, src_idx,
                                                               dst_idx, out, E_);
}
